// SphericalCodebook_25280177504373
// MI455X (gfx1250) — compile-verified, matrix operand order checked by reading, not run on hardware
//
#include <hip/hip_runtime.h>
#include <math.h>

// ---- problem constants --------------------------------------------------
#define B_ROWS   32768
#define K_DIM    512
#define N_CODES  8192
#define NT_TILES (N_CODES / 16)
#define TILE_BYTES (16 * K_DIM * 2)   // 16 KB per 16-code bf16 tile

typedef __attribute__((ext_vector_type(16))) __bf16 v16bf;
typedef __attribute__((ext_vector_type(8)))  __bf16 v8bf;
typedef __attribute__((ext_vector_type(8)))  float  v8f;
typedef __attribute__((ext_vector_type(4)))  float  v4f;

// exact pointee type expected by the async-LDS builtin (per hipcc diagnostic)
typedef int v4i_vs __attribute__((vector_size(16)));
typedef __attribute__((address_space(1))) v4i_vs* as1_v4i;
typedef __attribute__((address_space(3))) v4i_vs* as3_v4i;

#if __has_builtin(__builtin_amdgcn_global_load_async_to_lds_b128)
#define USE_ASYNC_LDS 1
#else
#define USE_ASYNC_LDS 0
#endif

__device__ __forceinline__ void wait_asynccnt0() {
#if __has_builtin(__builtin_amdgcn_s_wait_asynccnt)
    __builtin_amdgcn_s_wait_asynccnt(0);
#else
    asm volatile("s_wait_asynccnt 0x0" ::: "memory");
#endif
}

// ------------------------------------------------------------------------
// Kernel 1: zero the usage-count buffer (must be re-zeroed every launch).
// ------------------------------------------------------------------------
__global__ void zero_f32_kernel(float* __restrict__ p, int n) {
    int i = blockIdx.x * blockDim.x + threadIdx.x;
    if (i < n) p[i] = 0.0f;
}

// ------------------------------------------------------------------------
// Kernel 2: L2-normalize codebook rows, emit bf16 copy (one wave per row).
// ------------------------------------------------------------------------
__global__ __launch_bounds__(256) void emb_norm_kernel(
    const float* __restrict__ emb, __bf16* __restrict__ out)
{
    const int row  = (blockIdx.x * 256 + threadIdx.x) >> 5;   // 8192 rows
    const int lane = threadIdx.x & 31;
    const float* r = emb + (size_t)row * K_DIM + lane * 16;

    v4f x[4];
    float ss = 0.0f;
#pragma unroll
    for (int i = 0; i < 4; ++i) {
        x[i] = *(const v4f*)(r + i * 4);
#pragma unroll
        for (int j = 0; j < 4; ++j) ss += x[i][j] * x[i][j];
    }
#pragma unroll
    for (int s = 1; s < 32; s <<= 1) ss += __shfl_xor(ss, s, 32);
    const float inv = 1.0f / fmaxf(sqrtf(ss), 1e-12f);

    v8bf o0, o1;
#pragma unroll
    for (int j = 0; j < 4; ++j) {
        o0[j]     = (__bf16)(x[0][j] * inv);
        o0[j + 4] = (__bf16)(x[1][j] * inv);
        o1[j]     = (__bf16)(x[2][j] * inv);
        o1[j + 4] = (__bf16)(x[3][j] * inv);
    }
    __bf16* orow = out + (size_t)row * K_DIM + lane * 16;
    *(v8bf*)(orow)     = o0;
    *(v8bf*)(orow + 8) = o1;
}

// ------------------------------------------------------------------------
// Kernel 3: fused bf16-WMMA GEMM + row argmax.
//   Block = 256 threads (8 waves); each wave owns a 16-row A tile resident
//   in 128 VGPRs.  B tiles (16 codes x 512 bf16) double-buffer through LDS
//   via GLOBAL_LOAD_ASYNC_TO_LDS_B128 (ASYNCcnt) when available.  Argmax is
//   lane-local in the hot loop (8 cmp/sel per tile); the 16-lane butterfly
//   merge runs once at the end.
// ------------------------------------------------------------------------
__global__ __launch_bounds__(256) void gemm_argmax_kernel(
    const float*  __restrict__ z_e,      // 32768 x 512, raw f32
    const __bf16* __restrict__ emb_bf,   // 8192 x 512, normalized bf16
    int*   __restrict__ idx_out,         // 32768
    float* __restrict__ ind_f_out,       // 32768 (indices as float in d_out)
    float* __restrict__ maxs_out,        // 32768 (max cosine per row)
    float* __restrict__ counts)          // 8192
{
    __shared__ __align__(16) __bf16 lds_b[2][16 * K_DIM];   // 2 x 16 KB

    const int tid  = threadIdx.x;
    const int wave = tid >> 5;
    const int lane = tid & 31;
    const int m    = lane & 15;        // A-layout row / B-layout column
    const int hi   = lane >> 4;        // lane half selects K sub-pattern
    const int wave_row0 = blockIdx.x * 128 + wave * 16;

    // ---- prologue: load A tile as WMMA fragments, accumulate row sumsq --
    v16bf a[16];
    float sumsq = 0.0f;
    const float* arow = z_e + (size_t)(wave_row0 + m) * K_DIM;
#pragma unroll
    for (int kc = 0; kc < 16; ++kc) {
        const int k0 = kc * 32 + (hi ? 8 : 0);    // elements 0..7  = K k0..k0+7
        const int k1 = kc * 32 + (hi ? 24 : 16);  // elements 8..15 = K k1..k1+7
        v4f f0 = *(const v4f*)(arow + k0);
        v4f f1 = *(const v4f*)(arow + k0 + 4);
        v4f f2 = *(const v4f*)(arow + k1);
        v4f f3 = *(const v4f*)(arow + k1 + 4);
        v16bf av;
#pragma unroll
        for (int j = 0; j < 4; ++j) {
            av[j]      = (__bf16)f0[j];  sumsq += f0[j] * f0[j];
            av[j + 4]  = (__bf16)f1[j];  sumsq += f1[j] * f1[j];
            av[j + 8]  = (__bf16)f2[j];  sumsq += f2[j] * f2[j];
            av[j + 12] = (__bf16)f3[j];  sumsq += f3[j] * f3[j];
        }
        a[kc] = av;
    }
    // lanes m and m+16 jointly hold row m -> combine for full row norm
    sumsq += __shfl_xor(sumsq, 16, 32);
    const float inv_norm = 1.0f / fmaxf(sqrtf(sumsq), 1e-12f);

    // lane-local running argmax: this lane only ever sees codes nt*16 + m,
    // so tracking the best tile index nt is enough.
    float best[8];
    int   bnt[8];
#pragma unroll
    for (int r = 0; r < 8; ++r) { best[r] = -3.0e38f; bnt[r] = 0; }

#if USE_ASYNC_LDS
    // async staging: each thread issues 4 x b128 (16 KB / 256 threads)
    auto issue_tile = [&](int nt, int buf) {
        const char* g = (const char*)emb_bf + (size_t)nt * TILE_BYTES;
        char* l = (char*)&lds_b[buf][0];
#pragma unroll
        for (int i = 0; i < 4; ++i) {
            const int off = (tid + i * 256) * 16;
            __builtin_amdgcn_global_load_async_to_lds_b128(
                (as1_v4i)(uintptr_t)(g + off),
                (as3_v4i)(uintptr_t)(l + off),
                0, 0);
        }
    };
    issue_tile(0, 0);
#endif

    for (int nt = 0; nt < NT_TILES; ++nt) {
        const int cur = nt & 1;
#if USE_ASYNC_LDS
        wait_asynccnt0();        // my async slices of tile nt are in LDS
        __syncthreads();         // everyone's slices visible; prev reads done
        if (nt + 1 < NT_TILES) issue_tile(nt + 1, cur ^ 1);
#else
        __syncthreads();
        {   // synchronous staging fallback
            const uint4* gsrc = (const uint4*)((const char*)emb_bf +
                                               (size_t)nt * TILE_BYTES);
            uint4* ldst = (uint4*)&lds_b[cur][0];
#pragma unroll
            for (int i = 0; i < 4; ++i) ldst[tid + i * 256] = gsrc[tid + i * 256];
        }
        __syncthreads();
#endif
        const __bf16* bbase = &lds_b[cur][0];
        v8f c = {};
#pragma unroll
        for (int kc = 0; kc < 16; ++kc) {
            const int kb0 = kc * 32 + (hi ? 8 : 0);
            const int kb1 = kc * 32 + (hi ? 24 : 16);
            v8bf b0 = *(const v8bf*)(bbase + m * K_DIM + kb0);
            v8bf b1 = *(const v8bf*)(bbase + m * K_DIM + kb1);
            v16bf b = __builtin_shufflevector(b0, b1,
                0, 1, 2, 3, 4, 5, 6, 7, 8, 9, 10, 11, 12, 13, 14, 15);
            c = __builtin_amdgcn_wmma_f32_16x16x32_bf16(
                    false, a[kc], false, b, (short)0, c, false, false);
        }
        // strict '>' keeps the earliest (smallest) code on per-lane ties
#pragma unroll
        for (int r = 0; r < 8; ++r) {
            if (c[r] > best[r]) { best[r] = c[r]; bnt[r] = nt; }
        }
    }

    // ---- one-time cross-lane argmax merge within each 16-lane half -------
    int bidx[8];
#pragma unroll
    for (int r = 0; r < 8; ++r) {
        float v   = best[r];
        int  code = bnt[r] * 16 + m;
#pragma unroll
        for (int s = 1; s < 16; s <<= 1) {
            float ov = __shfl_xor(v, s, 32);
            int   oc = __shfl_xor(code, s, 32);
            if (ov > v || (ov == v && oc < code)) { v = ov; code = oc; }
        }
        best[r] = v; bidx[r] = code;
    }

    // ---- epilogue: fetch per-row inv_norm, emit results ------------------
    float invr[8];
#pragma unroll
    for (int r = 0; r < 8; ++r)
        invr[r] = __shfl(inv_norm, hi ? (r + 8) : r, 32);   // all lanes active

    if (m == 0) {           // lanes 0 and 16 cover rows r and r+8
#pragma unroll
        for (int r = 0; r < 8; ++r) {
            const int grow = wave_row0 + (hi ? (r + 8) : r);
            idx_out[grow]   = bidx[r];
            ind_f_out[grow] = (float)bidx[r];
            maxs_out[grow]  = best[r] * invr[r];   // true cosine for losses
            atomicAdd(&counts[bidx[r]], 1.0f);     // integer-valued: exact
        }
    }
}

// ------------------------------------------------------------------------
// Kernel 4: gather z_q = l2_normalize(emb[idx]) in f32 (one wave per row).
// ------------------------------------------------------------------------
__global__ __launch_bounds__(256) void gather_kernel(
    const float* __restrict__ emb, const int* __restrict__ idx,
    float* __restrict__ zq)
{
    const int row  = (blockIdx.x * 256 + threadIdx.x) >> 5;   // 32768 rows
    const int lane = threadIdx.x & 31;
    const int code = idx[row];
    const float* e = emb + (size_t)code * K_DIM + lane * 16;

    v4f x[4];
    float ss = 0.0f;
#pragma unroll
    for (int i = 0; i < 4; ++i) {
        x[i] = *(const v4f*)(e + i * 4);
#pragma unroll
        for (int j = 0; j < 4; ++j) ss += x[i][j] * x[i][j];
    }
#pragma unroll
    for (int s = 1; s < 32; s <<= 1) ss += __shfl_xor(ss, s, 32);
    const float inv = 1.0f / fmaxf(sqrtf(ss), 1e-12f);

    float* o = zq + (size_t)row * K_DIM + lane * 16;
#pragma unroll
    for (int i = 0; i < 4; ++i) {
        v4f y;
#pragma unroll
        for (int j = 0; j < 4; ++j) y[j] = x[i][j] * inv;
        *(v4f*)(o + i * 4) = y;
    }
}

// ------------------------------------------------------------------------
// Kernel 5: deterministic scalar stats.
//   losses: ||z_e_norm - z_q||^2 = 2 - 2*cos  (both unit norm), so
//   mean over (B,D) = (2 - 2*mean(max_cos)) / D.
// ------------------------------------------------------------------------
__global__ __launch_bounds__(256) void finalize_kernel(
    const float* __restrict__ counts, const float* __restrict__ maxs,
    float* __restrict__ out4)
{
    __shared__ float sh[3][8];
    float ent = 0.0f, used = 0.0f, ssum = 0.0f;
    for (int k = threadIdx.x; k < N_CODES; k += 256) {
        float avg = counts[k] * (1.0f / (float)B_ROWS);
        ent  -= avg * logf(avg + 1e-10f);
        used += (avg > 0.0f) ? 1.0f : 0.0f;
    }
    for (int i = threadIdx.x; i < B_ROWS; i += 256) ssum += maxs[i];
#pragma unroll
    for (int s = 1; s < 32; s <<= 1) {
        ent  += __shfl_xor(ent,  s, 32);
        used += __shfl_xor(used, s, 32);
        ssum += __shfl_xor(ssum, s, 32);
    }
    const int w = threadIdx.x >> 5, l = threadIdx.x & 31;
    if (l == 0) { sh[0][w] = ent; sh[1][w] = used; sh[2][w] = ssum; }
    __syncthreads();
    if (threadIdx.x == 0) {
        float e = 0.0f, u = 0.0f, s = 0.0f;
        for (int i = 0; i < 8; ++i) { e += sh[0][i]; u += sh[1][i]; s += sh[2][i]; }
        const float mean_s = s * (1.0f / (float)B_ROWS);
        const float md = (2.0f - 2.0f * mean_s) * (1.0f / (float)K_DIM);
        out4[0] = 0.25f * md;              // commitment_loss
        out4[1] = md;                      // codebook_loss
        out4[2] = expf(e);                 // perplexity
        out4[3] = u * (1.0f / (float)N_CODES);  // utilization
    }
}

// ------------------------------------------------------------------------
extern "C" void kernel_launch(void* const* d_in, const int* in_sizes, int n_in,
                              void* d_out, int out_size, void* d_ws, size_t ws_size,
                              hipStream_t stream)
{
    const float* z_e = (const float*)d_in[0];   // 32768 x 512
    const float* emb = (const float*)d_in[1];   // 8192 x 512

    float* out   = (float*)d_out;
    float* zq    = out;                                   // 32768*512
    float* ind_f = out + (size_t)B_ROWS * K_DIM;          // 32768
    float* scal  = ind_f + B_ROWS;                        // 4 scalars

    // workspace layout (~8.7 MB)
    char* ws = (char*)d_ws;
    __bf16* emb_bf  = (__bf16*)ws;                                    // 8 MB
    int*    idx     = (int*)  (ws + (size_t)N_CODES * K_DIM * 2);     // 128 KB
    float*  maxs    = (float*)((char*)idx + (size_t)B_ROWS * 4);      // 128 KB
    float*  counts  = maxs + B_ROWS;                                  // 32 KB

    zero_f32_kernel<<<(N_CODES + 255) / 256, 256, 0, stream>>>(counts, N_CODES);
    emb_norm_kernel<<<N_CODES / 8, 256, 0, stream>>>(emb, emb_bf);
    gemm_argmax_kernel<<<B_ROWS / 128, 256, 0, stream>>>(
        z_e, emb_bf, idx, ind_f, maxs, counts);
    gather_kernel<<<B_ROWS / 8, 256, 0, stream>>>(emb, idx, zq);
    finalize_kernel<<<1, 256, 0, stream>>>(counts, maxs, scal);
}
